// prealign_4647154614611
// MI455X (gfx1250) — matrix operand order repaired and checked by reading, not compiled, not run on hardware
//
#include <hip/hip_runtime.h>

#define NFC 64
#define HH 128
#define WW 128
#define BB 4

typedef __attribute__((ext_vector_type(16))) __bf16 v16bf;
typedef __attribute__((ext_vector_type(8)))  __bf16 v8bf;
typedef __attribute__((ext_vector_type(8)))  float  v8f;

__device__ __forceinline__ __bf16 f2bf(float f) {
    unsigned u = __builtin_bit_cast(unsigned, f);
    u += 0x7fffu + ((u >> 16) & 1u);            // round-to-nearest-even
    return __builtin_bit_cast(__bf16, (unsigned short)(u >> 16));
}
__device__ __forceinline__ float bf2f(__bf16 b) {
    unsigned u = (unsigned)__builtin_bit_cast(unsigned short, b) << 16;
    return __builtin_bit_cast(float, u);
}
__device__ __forceinline__ float lrelu(float v) { return v >= 0.f ? v : 0.1f * v; }
__device__ __forceinline__ v16bf cat8(v8bf a, v8bf b) {
    return __builtin_shufflevector(a, b, 0,1,2,3,4,5,6,7,8,9,10,11,12,13,14,15);
}

// ---------------------------------------------------------------------------
// NCHW f32 -> NHWC bf16 (optionally channel-concat of two 64-ch tensors)
// ---------------------------------------------------------------------------
template <int LOGC, bool CAT>
__global__ __launch_bounds__(256)
void pack_nhwc(const float* __restrict__ a, const float* __restrict__ b2,
               __bf16* __restrict__ dst)
{
    constexpr int C = 1 << LOGC;
    const size_t i = (size_t)blockIdx.x * 256 + threadIdx.x;
    if (i >= (size_t)BB * HH * WW * C) return;
    const int    c   = (int)i & (C - 1);
    const size_t pix = i >> LOGC;
    const int    b   = (int)(pix >> 14);          // H*W = 16384
    const size_t hw  = pix & 16383;
    const float* s; int cc;
    if (CAT) { s = (c < NFC) ? a : b2; cc = c & (NFC - 1); }
    else     { s = a; cc = c; }
    dst[i] = f2bf(s[((size_t)b * NFC + cc) * 16384 + hw]);
}

// ---------------------------------------------------------------------------
// OIHW f32 weights -> [tap r][n (padded w/ zeros to NPAD)][c] bf16
// ---------------------------------------------------------------------------
template <int LOGCIN, int COUT, int NPAD>
__global__ __launch_bounds__(256)
void wtrans(const float* __restrict__ w, __bf16* __restrict__ wt)
{
    constexpr int CIN = 1 << LOGCIN;
    const int i = blockIdx.x * 256 + threadIdx.x;
    if (i >= 9 * NPAD * CIN) return;
    const int c = i & (CIN - 1);
    const int t = i >> LOGCIN;
    const int n = t % NPAD;
    const int r = t / NPAD;
    const float v = (n < COUT) ? w[((size_t)n * CIN + c) * 9 + r] : 0.f;
    wt[i] = f2bf(v);
}

// w_dcn [n][c][r] f32 -> [n][k = r*64 + c] bf16  (matches sampled-S K order)
__global__ __launch_bounds__(256)
void wtrans_dcn(const float* __restrict__ w, __bf16* __restrict__ wt)
{
    const int i = blockIdx.x * 256 + threadIdx.x;
    if (i >= 64 * 576) return;
    const int k = i % 576;
    const int n = i / 576;
    const int r = k >> 6;
    const int c = k & 63;
    wt[i] = f2bf(w[((size_t)n * 64 + c) * 9 + r]);
}

// ---------------------------------------------------------------------------
// 3x3 SAME conv = 9 shifted 1x1 GEMMs on v_wmma_f32_16x16x32_bf16.
// One wave / block: 64 pixels (4 M-tiles) x 16 out channels; B reused 4x.
// in: NHWC bf16 [B][H][W][CIN]; wt: [9][NPAD][CIN] bf16.
// ---------------------------------------------------------------------------
template <int CIN, int COUT, int NPAD, bool LRELU, bool OUT_F32>
__global__ __launch_bounds__(32)
void conv3x3_nhwc_wmma(const __bf16* __restrict__ in, const __bf16* __restrict__ wt,
                       const float* __restrict__ bias, void* __restrict__ outv)
{
    const int lane = threadIdx.x;
    const int hi   = lane >> 4;
    const int m    = lane & 15;               // A row / pixel-in-tile, B col / channel
    const int blk  = blockIdx.x;              // B*H*(W/64)
    const int n0   = blockIdx.y * 16;

    const int x0 = (blk & (WW / 64 - 1)) * 64;
    const int y  = (blk >> 1) & (HH - 1);
    const int b  = blk >> 8;

    v8f acc[4] = {};
    v16bf zfrag;
#pragma unroll
    for (int q = 0; q < 16; ++q) zfrag[q] = __builtin_bit_cast(__bf16, (unsigned short)0);

#pragma unroll
    for (int r = 0; r < 9; ++r) {
        const int dy = r / 3 - 1, dx = r % 3 - 1;
        const int yy = y + dy;
        const bool rowok = (unsigned)yy < (unsigned)HH;
        const __bf16* rowp = in + (((size_t)b * HH + (rowok ? yy : 0)) * WW) * CIN;
#pragma unroll
        for (int c0 = 0; c0 < CIN; c0 += 32) {
            const __bf16* bp = wt + ((size_t)r * NPAD + n0 + m) * CIN + c0 + (hi << 3);
            const v16bf bfrag = cat8(*(const v8bf*)bp, *(const v8bf*)(bp + 16));
#pragma unroll
            for (int s = 0; s < 4; ++s) {
                const int xx = x0 + s * 16 + m + dx;
                v16bf afrag;
                if (rowok && (unsigned)xx < (unsigned)WW) {
                    const __bf16* p = rowp + (size_t)xx * CIN + c0 + (hi << 3);
                    afrag = cat8(*(const v8bf*)p, *(const v8bf*)(p + 16));
                } else {
                    afrag = zfrag;
                }
                acc[s] = __builtin_amdgcn_wmma_f32_16x16x32_bf16(
                    false, afrag, false, bfrag, (short)0, acc[s], false, false);
            }
        }
    }

    const int nch = n0 + m;
    if (nch < COUT) {
        const float bv = bias[nch];
#pragma unroll
        for (int s = 0; s < 4; ++s) {
#pragma unroll
            for (int rr = 0; rr < 8; ++rr) {
                float v = acc[s][rr] + bv;
                if (LRELU) v = lrelu(v);
                const int xx = x0 + s * 16 + rr + (hi << 3);
                const size_t o = (((size_t)b * HH + y) * WW + xx) * COUT + nch;
                if (OUT_F32) ((float*)outv)[o] = v;
                else         ((__bf16*)outv)[o] = f2bf(v);
            }
        }
    }
}

// ---------------------------------------------------------------------------
// DCNv2 sampling: S[pix][r*64 + g*8 + cl] = mask * bilinear(x, p+offset)
// com NHWC f32 [B][H][W][216]: ch gr = oy, 72+gr = ox, 144+gr = mask logit.
// One thread per (pixel, g, r): 4 taps x 8 channels, one 16B store.
// ---------------------------------------------------------------------------
__global__ __launch_bounds__(256)
void dcn_sample(const __bf16* __restrict__ xn, const float* __restrict__ com,
                __bf16* __restrict__ S)
{
    const size_t i = (size_t)blockIdx.x * 256 + threadIdx.x;
    if (i >= (size_t)BB * HH * WW * 72) return;
    const int    gr  = (int)(i % 72);
    const size_t pix = i / 72;
    const int g = gr / 9;
    const int r = gr - 9 * g;
    const int x = (int)(pix & (WW - 1));
    const int y = (int)((pix >> 7) & (HH - 1));
    const int b = (int)(pix >> 14);

    const float* cp = com + pix * 216;
    const float oy = cp[gr], ox = cp[72 + gr], mv = cp[144 + gr];
    const float mk = 1.f / (1.f + __expf(-mv));

    const float py  = (float)(y + r / 3 - 1) + oy;
    const float pxf = (float)(x + r % 3 - 1) + ox;
    const float y0f = floorf(py), x0f = floorf(pxf);
    const float wy = py - y0f, wx = pxf - x0f;
    const int yi = (int)y0f, xi = (int)x0f;
    const float w00 = mk * (1.f - wy) * (1.f - wx);
    const float w01 = mk * (1.f - wy) * wx;
    const float w10 = mk * wy * (1.f - wx);
    const float w11 = mk * wy * wx;

    const __bf16* xb = xn + ((size_t)b << 14) * 64 + g * 8;
    float res[8] = {0.f, 0.f, 0.f, 0.f, 0.f, 0.f, 0.f, 0.f};
    auto tap = [&](int ty, int tx, float wgt) {
        if ((unsigned)ty < (unsigned)HH && (unsigned)tx < (unsigned)WW) {
            const v8bf v = *(const v8bf*)(xb + (((size_t)ty << 7) + tx) * 64);
#pragma unroll
            for (int c = 0; c < 8; ++c) res[c] += wgt * bf2f(v[c]);
        }
    };
    tap(yi,     xi,     w00);
    tap(yi,     xi + 1, w01);
    tap(yi + 1, xi,     w10);
    tap(yi + 1, xi + 1, w11);

    v8bf ov;
#pragma unroll
    for (int c = 0; c < 8; ++c) ov[c] = f2bf(res[c]);
    *(v8bf*)(S + pix * 576 + r * 64 + g * 8) = ov;
}

// ---------------------------------------------------------------------------
// DCN einsum as pure GEMM: M = pixels, N = 64, K = 576 (contiguous rows of S).
// Output written NCHW f32 (the reference output layout), with leaky-ReLU.
// ---------------------------------------------------------------------------
__global__ __launch_bounds__(32)
void dcn_gemm(const __bf16* __restrict__ S, const __bf16* __restrict__ wt,
              const float* __restrict__ bias, float* __restrict__ out)
{
    const int lane = threadIdx.x;
    const int hi   = lane >> 4;
    const int m    = lane & 15;
    const int blk  = blockIdx.x;              // B*H*W/64 pixel blocks
    const int n0   = blockIdx.y * 16;

    const size_t p0     = (size_t)blk * 64;
    const int    b      = blk >> 8;           // 16384/64 = 256 blocks per image
    const int    hwbase = (blk & 255) * 64;

    v8f acc[4] = {};
#pragma unroll
    for (int k0 = 0; k0 < 576; k0 += 32) {
        const __bf16* bp = wt + (size_t)(n0 + m) * 576 + k0 + (hi << 3);
        const v16bf bfrag = cat8(*(const v8bf*)bp, *(const v8bf*)(bp + 16));
#pragma unroll
        for (int s = 0; s < 4; ++s) {
            const __bf16* p = S + (p0 + s * 16 + m) * 576 + k0 + (hi << 3);
            const v16bf afrag = cat8(*(const v8bf*)p, *(const v8bf*)(p + 16));
            acc[s] = __builtin_amdgcn_wmma_f32_16x16x32_bf16(
                false, afrag, false, bfrag, (short)0, acc[s], false, false);
        }
    }

    const int nch = n0 + m;
    const float bv = bias[nch];
#pragma unroll
    for (int s = 0; s < 4; ++s) {
#pragma unroll
        for (int rr = 0; rr < 8; ++rr) {
            const int hw = hwbase + s * 16 + rr + (hi << 3);
            out[((size_t)b * NFC + nch) * 16384 + hw] = lrelu(acc[s][rr] + bv);
        }
    }
}

extern "C" void kernel_launch(void* const* d_in, const int* in_sizes, int n_in,
                              void* d_out, int out_size, void* d_ws, size_t ws_size,
                              hipStream_t stream)
{
    const float* angular = (const float*)d_in[0];
    const float* center  = (const float*)d_in[1];
    const float* w_off1  = (const float*)d_in[2];
    const float* b_off1  = (const float*)d_in[3];
    const float* w_off2  = (const float*)d_in[4];
    const float* b_off2  = (const float*)d_in[5];
    const float* w_com   = (const float*)d_in[6];
    const float* b_com   = (const float*)d_in[7];
    const float* w_dcn   = (const float*)d_in[8];
    const float* b_dcn   = (const float*)d_in[9];

    char* ws = (char*)d_ws;
    // S (75.5 MB) region is dead until dcn_sample; alias conv temporaries in it.
    __bf16* S      = (__bf16*)ws;                         // [B*H*W][576]
    __bf16* cat_in = (__bf16*)ws;                         // [B*H*W][128] 16.8 MB
    __bf16* t1     = (__bf16*)(ws + 16777216);            // [B*H*W][64]   8.4 MB
    __bf16* t2     = (__bf16*)(ws + 25165824);            // [B*H*W][64]   8.4 MB
    size_t off = 75497472;
    __bf16* xn  = (__bf16*)(ws + off); off += 8388608;    // x NHWC bf16
    float*  com = (float*) (ws + off); off += 56623104;   // [B*H*W][216] f32
    __bf16* wt1 = (__bf16*)(ws + off); off += 147456;     // [9][64][128]
    __bf16* wt2 = (__bf16*)(ws + off); off += 73728;      // [9][64][64]
    __bf16* wt3 = (__bf16*)(ws + off); off += 258048;     // [9][224][64]
    __bf16* wtd = (__bf16*)(ws + off); off += 73728;      // [64][576]

    const int PIX = BB * HH * WW;                         // 65536

    pack_nhwc<7, true ><<<(PIX * 128 + 255) / 256, 256, 0, stream>>>(angular, center, cat_in);
    pack_nhwc<6, false><<<(PIX * 64  + 255) / 256, 256, 0, stream>>>(angular, angular, xn);
    wtrans<7, 64, 64  ><<<(9 * 64 * 128 + 255) / 256, 256, 0, stream>>>(w_off1, wt1);
    wtrans<6, 64, 64  ><<<(9 * 64 * 64  + 255) / 256, 256, 0, stream>>>(w_off2, wt2);
    wtrans<6, 216, 224><<<(9 * 224 * 64 + 255) / 256, 256, 0, stream>>>(w_com, wt3);
    wtrans_dcn<<<(64 * 576 + 255) / 256, 256, 0, stream>>>(w_dcn, wtd);

    const dim3 cgrid(BB * HH * (WW / 64), 4);
    conv3x3_nhwc_wmma<128, 64, 64, true,  false><<<cgrid, 32, 0, stream>>>(cat_in, wt1, b_off1, t1);
    conv3x3_nhwc_wmma<64,  64, 64, true,  false><<<cgrid, 32, 0, stream>>>(t1, wt2, b_off2, t2);
    conv3x3_nhwc_wmma<64, 216, 224, false, true><<<dim3(BB * HH * 2, 14), 32, 0, stream>>>(t2, wt3, b_com, com);

    dcn_sample<<<((size_t)PIX * 72 + 255) / 256, 256, 0, stream>>>(xn, com, S);
    dcn_gemm<<<dim3(PIX / 64, 4), 32, 0, stream>>>(S, wtd, b_dcn, (float*)d_out);
}
